// cut_MPNN_7481833029837
// MI455X (gfx1250) — matrix operand order repaired and checked by reading, not compiled, not run on hardware
//
#include <hip/hip_runtime.h>
#include <hip/hip_bf16.h>
#include <stdint.h>

// ---------------------------------------------------------------------------
// cut_MPNN for MI455X (gfx1250, wave32).
//
// Dense 64x64 layers -> v_wmma_f32_16x16x32_f16 (fp32 accumulate). Weights are
// staged TRANSPOSED in LDS with a 72-half row stride, so every B fragment is
// two contiguous 16B ds_load_b128 per lane, bank-conflict-free (36c mod 64 is
// a permutation over 16 columns). Edge segment ops -> global float atomics
// (all [N,64] buffers fit in the 192MB L2, so atomic RMW stays on-chip).
// Stream-once GAT edge weights use non-temporal TH hints.
// Input pointer layout assumes jax pytree flattening of setup_inputs():
//   0: x[N]
//   1: bn2.b  2: bn2.g
//   3..8:  bns[i].{b,g} for i=0..2
//   9: conv1.b 10: conv1.g 11: conv1.l1.W[1,64] 12: conv1.l1.b
//   13: conv1.l2.W[64,64] 14: conv1.l2.b
//   15+6i .. 20+6i: convs[i].{b,g,l1.W,l1.b,l2.W,l2.b} for i=0..2
//   33: gat.W[64,64] 34: gat.att_dst[8,8] 35: gat.att_src[8,8] 36: gat.b[64]
//   37: lin1.W 38: lin1.b 39: lin2.W[64,1] 40: lin2.b[1]
//   41: edge_index[2,E] int32 (src then dst)   42: batch[N] int32
// d_out: probs[N] | expected_cut[G] | vol_hard[G] | tv_hard[G]
// ---------------------------------------------------------------------------

typedef __attribute__((ext_vector_type(16))) _Float16 v16h;
typedef __attribute__((ext_vector_type(8)))  _Float16 vh8;
typedef __attribute__((ext_vector_type(8)))  float    v8f;

#define GG 256      // graphs
#define FD 64       // H1 = HEADS*H2 = 64
#define NHEAD 8
#define WSTRIDE 72  // padded LDS row stride (halfs): conflict-free B reads

__device__ __forceinline__ float leakyf(float v, float s) { return v > 0.f ? v : s * v; }

__device__ __forceinline__ void atomicMaxF(float* a, float v) {
  int old = __float_as_int(*a);
  while (__int_as_float(old) < v) {
    int assumed = old;
    old = atomicCAS((int*)a, assumed, __float_as_int(v));
    if (old == assumed) break;
  }
}
__device__ __forceinline__ void atomicMinF(float* a, float v) {
  int old = __float_as_int(*a);
  while (__int_as_float(old) > v) {
    int assumed = old;
    old = atomicCAS((int*)a, assumed, __float_as_int(v));
    if (old == assumed) break;
  }
}

// deterministic per-element uniform stand-in for jax.random.uniform(key(1))
__device__ __forceinline__ float u01(uint32_t i) {
  uint32_t s = i * 747796405u + 2891336453u;
  s = ((s >> ((s >> 28) + 4u)) ^ s) * 277803737u;
  s = (s >> 22) ^ s;
  return (float)(s >> 8) * (1.0f / 16777216.0f);
}

// ------------------------------ WMMA GEMM ----------------------------------
// Y[n,64] = act(X[n,64] @ W[64,64] + bias)   ACT: 0=none 1=relu 2=leaky(.01)
// Block = 256 threads = 8 waves; each wave computes one 16-row tile x 64 cols
// via 4 column tiles x 2 K-steps = 8 v_wmma_f32_16x16x32_f16.
template <int ACT, bool HAS_BIAS>
__global__ __launch_bounds__(256) void gemm_n64_wmma_kernel(
    const float* __restrict__ X, const float* __restrict__ W,
    const float* __restrict__ bias, float* __restrict__ Y, int n) {
  __shared__ _Float16 sWT[FD * WSTRIDE];   // sWT[c*WSTRIDE + k] = W[k][c]
  const int tid = threadIdx.x;
  for (int i = tid; i < FD * FD; i += 256) {
    const int k = i >> 6, c = i & 63;
    sWT[c * WSTRIDE + k] = (_Float16)W[i];
  }
  __syncthreads();

  const int wave = tid >> 5;
  const int lane = tid & 31;
  const int l  = lane & 15;
  const int hi = lane >> 4;
  const int row0 = (blockIdx.x * 8 + wave) * 16;
  if (row0 >= n) return;                      // wave-uniform exit

  int arow = row0 + l;                        // A frag row = lane&15 (ISA layout)
  if (arow > n - 1) arow = n - 1;             // clamp keeps EXEC all-ones
  const float* xr = X + (size_t)arow * FD;

  // 16-bit A 16x32 K order chosen here (not ISA table order): lane group hi takes K+16 subrange; one contiguous
  // run of 16 per lane: [kb+hi*16, +16) (matches B fragment K order).
  v16h a0, a1;
#pragma unroll
  for (int t = 0; t < 8; ++t) {
    a0[t]     = (_Float16)xr[ 0 + hi * 16 + t];
    a0[8 + t] = (_Float16)xr[ 8 + hi * 16 + t];
    a1[t]     = (_Float16)xr[32 + hi * 16 + t];
    a1[8 + t] = (_Float16)xr[40 + hi * 16 + t];
  }

  const bool full = (row0 + 16 <= n);         // wave-uniform fast path

#pragma unroll
  for (int jb = 0; jb < 4; ++jb) {
    const int col = jb * 16 + l;              // B frag: lane&15 = N column
    // B frag K values are contiguous in transposed LDS: 2x 16B vector loads.
    const vh8* bp0 = (const vh8*)&sWT[col * WSTRIDE +  0 + hi * 16];
    const vh8* bp1 = (const vh8*)&sWT[col * WSTRIDE + 32 + hi * 16];
    const vh8 q0 = bp0[0], q1 = bp0[1];
    const vh8 q2 = bp1[0], q3 = bp1[1];
    v16h b0, b1;
#pragma unroll
    for (int t = 0; t < 8; ++t) {
      b0[t] = q0[t]; b0[8 + t] = q1[t];
      b1[t] = q2[t]; b1[8 + t] = q3[t];
    }
    const float bv = HAS_BIAS ? bias[col] : 0.0f;
    v8f c;
#pragma unroll
    for (int r = 0; r < 8; ++r) c[r] = bv;    // C[m, l]: bias depends on N only
    c = __builtin_amdgcn_wmma_f32_16x16x32_f16(false, a0, false, b0, (short)0, c, false, false);
    c = __builtin_amdgcn_wmma_f32_16x16x32_f16(false, a1, false, b1, (short)0, c, false, false);
#pragma unroll
    for (int r = 0; r < 8; ++r) {
      float v = c[r];
      if (ACT == 1) v = v > 0.f ? v : 0.f;
      else if (ACT == 2) v = leakyf(v, 0.01f);
      c[r] = v;
    }
    float* yr = Y + (size_t)(row0 + hi * 8) * FD + col;
    if (full) {
#pragma unroll
      for (int r = 0; r < 8; ++r) yr[(size_t)r * FD] = c[r];   // D: M = r + 8*hi
    } else {
#pragma unroll
      for (int r = 0; r < 8; ++r)
        if (row0 + hi * 8 + r < n) yr[(size_t)r * FD] = c[r];
    }
  }
}

// --------------------------- elementwise helpers ---------------------------
__global__ void fill_kernel(float* p, float v, long long cnt) {
  long long i = blockIdx.x * (long long)blockDim.x + threadIdx.x;
  if (i < cnt) p[i] = v;
}
__global__ void binmask_kernel(const float* __restrict__ x, float* __restrict__ m, int n) {
  int i = blockIdx.x * blockDim.x + threadIdx.x;
  if (i < n) m[i] = (x[i] != 0.f) ? 1.f : 0.f;
}
__global__ void maskprop_kernel(const float* __restrict__ m, const int* __restrict__ src,
                                const int* __restrict__ dst, float* __restrict__ m2, int e) {
  int i = blockIdx.x * blockDim.x + threadIdx.x;
  if (i < e && m[src[i]] != 0.f)
    atomicMax((int*)&m2[dst[i]], __float_as_int(1.0f));  // monotonic for >=0 floats
}
// float4 over [N,64]: i in [0, n*16)
__global__ void maskmul_kernel(const float4* __restrict__ X, const float* __restrict__ m,
                               float4* __restrict__ Y, long long n16) {
  long long i = blockIdx.x * (long long)blockDim.x + threadIdx.x;
  if (i >= n16) return;
  const float mk = m[i >> 4];
  float4 v = X[i];
  v.x *= mk; v.y *= mk; v.z *= mk; v.w *= mk;
  Y[i] = v;
}
__global__ void add_kernel(float4* __restrict__ A, const float4* __restrict__ B, long long cnt4) {
  long long i = blockIdx.x * (long long)blockDim.x + threadIdx.x;
  if (i >= cnt4) return;
  float4 a = A[i]; const float4 b = B[i];
  a.x += b.x; a.y += b.y; a.z += b.z; a.w += b.w;
  A[i] = a;
}

// ----------------------------- segment scatters ----------------------------
__global__ void scatter1_kernel(const float* __restrict__ h, const int* __restrict__ src,
                                const int* __restrict__ dst, float* __restrict__ z, int e) {
  int i = blockIdx.x * blockDim.x + threadIdx.x;
  if (i < e) atomicAdd(&z[dst[i]], h[src[i]]);
}
// thread = (edge, 4-feature group): one float4 gather + 4 atomics
__global__ void scatter64_kernel(const float* __restrict__ h, const int* __restrict__ src,
                                 const int* __restrict__ dst, float* __restrict__ z, int e) {
  long long i = blockIdx.x * (long long)blockDim.x + threadIdx.x;
  if (i >= (long long)e * 16) return;
  const int ed = (int)(i >> 4), f = ((int)i & 15) * 4;
  const float4 v = *(const float4*)&h[(size_t)src[ed] * FD + f];
  float* zp = &z[(size_t)dst[ed] * FD + f];
  atomicAdd(&zp[0], v.x); atomicAdd(&zp[1], v.y);
  atomicAdd(&zp[2], v.z); atomicAdd(&zp[3], v.w);
}

// conv1 first linear (fin=1): y[n,f] = relu(z[n]*W[f] + b[f]), float4-wide
__global__ void expand1_kernel(const float* __restrict__ z1, const float* __restrict__ W,
                               const float* __restrict__ b, float4* __restrict__ y, long long n16) {
  long long i = blockIdx.x * (long long)blockDim.x + threadIdx.x;
  if (i >= n16) return;
  const int f = ((int)i & 15) * 4;
  const float zv = z1[i >> 4];
  float4 v;
  v.x = fmaxf(zv * W[f + 0] + b[f + 0], 0.f);
  v.y = fmaxf(zv * W[f + 1] + b[f + 1], 0.f);
  v.z = fmaxf(zv * W[f + 2] + b[f + 2], 0.f);
  v.w = fmaxf(zv * W[f + 3] + b[f + 3], 0.f);
  y[i] = v;
}

// ------------------------------- batch norm --------------------------------
// Coalesced float4 partial sums; launch <<<64,256>>> so the grid stride
// (16384) is a multiple of 16: each thread owns a fixed 4-column group.
__global__ __launch_bounds__(256) void bn_partial_kernel(const float4* __restrict__ X,
                                                         long long n16,
                                                         float* __restrict__ sums,
                                                         float* __restrict__ sqs) {
  const long long i0 = blockIdx.x * 256LL + threadIdx.x;
  const long long stride = 64LL * 256LL;
  float s0 = 0.f, s1 = 0.f, s2 = 0.f, s3 = 0.f;
  float q0 = 0.f, q1 = 0.f, q2 = 0.f, q3 = 0.f;
  for (long long i = i0; i < n16; i += stride) {
    const float4 v = X[i];
    s0 += v.x; q0 += v.x * v.x;
    s1 += v.y; q1 += v.y * v.y;
    s2 += v.z; q2 += v.z * v.z;
    s3 += v.w; q3 += v.w * v.w;
  }
  const int f = ((int)(i0 & 15)) * 4;
  atomicAdd(&sums[f + 0], s0); atomicAdd(&sqs[f + 0], q0);
  atomicAdd(&sums[f + 1], s1); atomicAdd(&sqs[f + 1], q1);
  atomicAdd(&sums[f + 2], s2); atomicAdd(&sqs[f + 2], q2);
  atomicAdd(&sums[f + 3], s3); atomicAdd(&sqs[f + 3], q3);
}
__global__ void bn_finalize_kernel(const float* __restrict__ sums, const float* __restrict__ sqs,
                                   int n, float* __restrict__ mean, float* __restrict__ var) {
  const int c = threadIdx.x;  // 64 threads
  const float m = sums[c] / n;
  mean[c] = m;
  var[c] = sqs[c] / n - m * m;
}
__global__ void bn_apply_kernel(float4* __restrict__ X, long long n16,
                                const float* __restrict__ mean, const float* __restrict__ var,
                                const float* __restrict__ g, const float* __restrict__ b) {
  long long i = blockIdx.x * (long long)blockDim.x + threadIdx.x;
  if (i >= n16) return;
  const int f = ((int)i & 15) * 4;
  float4 v = X[i];
  v.x = g[f + 0] * (v.x - mean[f + 0]) * rsqrtf(var[f + 0] + 1e-5f) + b[f + 0];
  v.y = g[f + 1] * (v.y - mean[f + 1]) * rsqrtf(var[f + 1] + 1e-5f) + b[f + 1];
  v.z = g[f + 2] * (v.z - mean[f + 2]) * rsqrtf(var[f + 2] + 1e-5f) + b[f + 2];
  v.w = g[f + 3] * (v.w - mean[f + 3]) * rsqrtf(var[f + 3] + 1e-5f) + b[f + 3];
  X[i] = v;
}

// ---------------------------------- GAT ------------------------------------
__global__ void attn_score_kernel(const float* __restrict__ hh, const float* __restrict__ asrc,
                                  const float* __restrict__ adst, float* __restrict__ a_s,
                                  float* __restrict__ a_d, int n) {
  int i = blockIdx.x * blockDim.x + threadIdx.x;
  if (i >= n) return;
  const float* r = hh + (size_t)i * FD;
#pragma unroll
  for (int h = 0; h < NHEAD; ++h) {
    float s = 0.f, d = 0.f;
#pragma unroll
    for (int f = 0; f < 8; ++f) {
      float v = r[h * 8 + f];
      s += v * asrc[h * 8 + f];
      d += v * adst[h * 8 + f];
    }
    a_s[i * NHEAD + h] = s;
    a_d[i * NHEAD + h] = d;
  }
}
// wexp is written once / read once: non-temporal TH hints keep the resident
// [N,64] working set in L2 instead of this 25.6MB stream.
__global__ void gat_edge1_kernel(const float* __restrict__ a_s, const float* __restrict__ a_d,
                                 const int* __restrict__ src, const int* __restrict__ dst,
                                 float* __restrict__ wexp, float* __restrict__ mx, int e) {
  long long i = blockIdx.x * (long long)blockDim.x + threadIdx.x;
  if (i >= (long long)e * NHEAD) return;
  const int ed = (int)(i >> 3), h = (int)(i & 7);
  float v = leakyf(a_s[src[ed] * NHEAD + h] + a_d[dst[ed] * NHEAD + h], 0.2f);
  __builtin_nontemporal_store(v, &wexp[i]);
  atomicMaxF(&mx[dst[ed] * NHEAD + h], v);
}
__global__ void gat_edge2_kernel(float* __restrict__ wexp, const float* __restrict__ mx,
                                 const int* __restrict__ dst, float* __restrict__ den, int e) {
  long long i = blockIdx.x * (long long)blockDim.x + threadIdx.x;
  if (i >= (long long)e * NHEAD) return;
  const int ed = (int)(i >> 3), h = (int)(i & 7);
  float w = __expf(__builtin_nontemporal_load(&wexp[i]) - mx[dst[ed] * NHEAD + h]);
  __builtin_nontemporal_store(w, &wexp[i]);
  atomicAdd(&den[dst[ed] * NHEAD + h], w);
}
// thread = (edge, 4-feature group); head constant within the group (4 | 8)
__global__ void gat_edge3_kernel(const float* __restrict__ wexp, const float* __restrict__ den,
                                 const float* __restrict__ hh, const int* __restrict__ src,
                                 const int* __restrict__ dst, float* __restrict__ out, int e) {
  long long i = blockIdx.x * (long long)blockDim.x + threadIdx.x;
  if (i >= (long long)e * 16) return;
  const int ed = (int)(i >> 4), f = ((int)i & 15) * 4, h = f >> 3;
  const int d = dst[ed];
  const float co = __builtin_nontemporal_load(&wexp[(size_t)ed * NHEAD + h]) /
                   (den[d * NHEAD + h] + 1e-16f);
  const float4 v = *(const float4*)&hh[(size_t)src[ed] * FD + f];
  float* op = &out[(size_t)d * FD + f];
  atomicAdd(&op[0], co * v.x); atomicAdd(&op[1], co * v.y);
  atomicAdd(&op[2], co * v.z); atomicAdd(&op[3], co * v.w);
}
__global__ void gat_finish_kernel(const float4* __restrict__ acc, const float* __restrict__ b,
                                  const float* __restrict__ m, float4* __restrict__ h,
                                  long long n16) {
  long long i = blockIdx.x * (long long)blockDim.x + threadIdx.x;
  if (i >= n16) return;
  const int f = ((int)i & 15) * 4;
  const float mk = m[i >> 4];
  float4 v = acc[i];
  v.x = (v.x + b[f + 0]) * mk;
  v.y = (v.y + b[f + 1]) * mk;
  v.z = (v.z + b[f + 2]) * mk;
  v.w = (v.w + b[f + 3]) * mk;
  h[i] = v;
}

// ------------------------------- output head -------------------------------
__global__ void lin2_kernel(const float* __restrict__ H, const float* __restrict__ W,
                            const float* __restrict__ b, const float* __restrict__ mask,
                            float* __restrict__ out, int n) {
  int i = blockIdx.x * blockDim.x + threadIdx.x;
  if (i >= n) return;
  float s = b[0];
  const float* r = H + (size_t)i * FD;
#pragma unroll
  for (int f = 0; f < FD; ++f) s += r[f] * W[f];
  out[i] = leakyf(s, 0.01f) * mask[i];
}
__global__ void minmax_kernel(const float* __restrict__ hs, const int* __restrict__ batch,
                              float* __restrict__ bmax, float* __restrict__ bmin, int n) {
  int i = blockIdx.x * blockDim.x + threadIdx.x;
  if (i >= n) return;
  atomicMaxF(&bmax[batch[i]], hs[i]);
  atomicMinF(&bmin[batch[i]], hs[i]);
}
__global__ void norm_kernel(float* __restrict__ hs, const float* __restrict__ x,
                            const float* __restrict__ m, const int* __restrict__ batch,
                            const float* __restrict__ bmax, const float* __restrict__ bmin, int n) {
  int i = blockIdx.x * blockDim.x + threadIdx.x;
  if (i >= n) return;
  const int g = batch[i];
  float hv = (hs[i] - bmin[g]) / (bmax[g] + 1e-6f - bmin[g]);
  float mk = m[i];
  hs[i] = hv * mk + mk * 1e-6f + x[i];
}
__global__ void deg_kernel(const int* __restrict__ src, float* __restrict__ deg, int e) {
  int i = blockIdx.x * blockDim.x + threadIdx.x;
  if (i < e) atomicAdd(&deg[src[i]], 1.0f);
}
__global__ void vol_kernel(const float* __restrict__ deg, const int* __restrict__ batch,
                           float* __restrict__ tvol, int n) {
  int i = blockIdx.x * blockDim.x + threadIdx.x;
  if (i < n) atomicAdd(&tvol[batch[i]], deg[i]);
}
__global__ void target_kernel(const float* __restrict__ tvol, float* __restrict__ target, int g) {
  int i = blockIdx.x * blockDim.x + threadIdx.x;
  if (i < g) target[i] = 0.3f * (tvol[i] + 1e-6f);
}
__global__ void refine_node_kernel(const float* __restrict__ hs, const float* __restrict__ deg,
                                   const float* __restrict__ m, const int* __restrict__ batch,
                                   const float* __restrict__ aG, float* __restrict__ sdnk,
                                   float* __restrict__ sxk, int n) {
  int i = blockIdx.x * blockDim.x + threadIdx.x;
  if (i >= n) return;
  const int g = batch[i];
  const float hv = hs[i], mk = m[i], d = deg[i];
  const float keep = (aG[g] * hv < 1.0f) ? 1.f : 0.f;
  atomicAdd(&sdnk[g], d * (1.f - keep) * mk);
  atomicAdd(&sxk[g], (hv * keep * mk) * (d * keep * mk));
}
__global__ void refine_update_kernel(float* __restrict__ aG, const float* __restrict__ target,
                                     const float* __restrict__ sdnk, const float* __restrict__ sxk,
                                     int g) {
  int i = blockIdx.x * blockDim.x + threadIdx.x;
  if (i < g) aG[i] = (target[i] - sdnk[i]) / (sxk[i] + 1e-5f);
}
__global__ void probs_kernel(const float* __restrict__ hs, const float* __restrict__ m,
                             const int* __restrict__ batch, const float* __restrict__ aG,
                             float* __restrict__ probs, float* __restrict__ x2, int n) {
  int i = blockIdx.x * blockDim.x + threadIdx.x;
  if (i >= n) return;
  float p = aG[batch[i]] * hs[i] * m[i];
  p = fminf(fmaxf(p, 0.f), 1.f);
  probs[i] = p;
  x2[i] = (p - u01((uint32_t)i) > 0.f) ? 1.f : 0.f;
}
__global__ void node_accum_kernel(const float* __restrict__ probs, const float* __restrict__ deg,
                                  const float* __restrict__ x2, const int* __restrict__ batch,
                                  float* __restrict__ vh, float* __restrict__ ec1, int n) {
  int i = blockIdx.x * blockDim.x + threadIdx.x;
  if (i >= n) return;
  const int g = batch[i];
  atomicAdd(&vh[g], deg[i] * x2[i]);
  atomicAdd(&ec1[g], probs[i] * deg[i]);
}
__global__ void edge_accum_kernel(const float* __restrict__ probs, const float* __restrict__ x2,
                                  const int* __restrict__ src, const int* __restrict__ dst,
                                  const int* __restrict__ batch, float* __restrict__ tvh,
                                  float* __restrict__ ec2, int e) {
  int i = blockIdx.x * blockDim.x + threadIdx.x;
  if (i >= e) return;
  const int s = src[i], d = dst[i], gs = batch[s];
  atomicAdd(&tvh[gs], 0.5f * fabsf(x2[s] - x2[d]));
  atomicAdd(&ec2[gs], probs[s] * probs[d]);
}
__global__ void finalize_kernel(const float* __restrict__ ec1, const float* __restrict__ ec2,
                                const float* __restrict__ vh, const float* __restrict__ tvh,
                                float* __restrict__ oec, float* __restrict__ ovh,
                                float* __restrict__ otv, int g) {
  int i = blockIdx.x * blockDim.x + threadIdx.x;
  if (i >= g) return;
  oec[i] = ec1[i] - ec2[i];
  ovh[i] = vh[i] + 1e-6f;
  otv[i] = tvh[i];
}

// ---------------------------------------------------------------------------
extern "C" void kernel_launch(void* const* d_in, const int* in_sizes, int n_in,
                              void* d_out, int out_size, void* d_ws, size_t ws_size,
                              hipStream_t stream) {
  (void)out_size; (void)ws_size;
  if (n_in < 43) return;

  const int N = in_sizes[0];
  const int E = in_sizes[41] / 2;
  const int G = GG;

  const float* x = (const float*)d_in[0];
  const float* bn2_b = (const float*)d_in[1];
  const float* bn2_g = (const float*)d_in[2];
  const float *bns_b[3], *bns_g[3];
  for (int i = 0; i < 3; ++i) { bns_b[i] = (const float*)d_in[3 + 2*i]; bns_g[i] = (const float*)d_in[4 + 2*i]; }
  const float* c1_b   = (const float*)d_in[9];
  const float* c1_g   = (const float*)d_in[10];
  const float* c1_l1W = (const float*)d_in[11];
  const float* c1_l1b = (const float*)d_in[12];
  const float* c1_l2W = (const float*)d_in[13];
  const float* c1_l2b = (const float*)d_in[14];
  const float *cv_b[3], *cv_g[3], *cv_l1W[3], *cv_l1b[3], *cv_l2W[3], *cv_l2b[3];
  for (int i = 0; i < 3; ++i) {
    int base = 15 + 6*i;
    cv_b[i]   = (const float*)d_in[base + 0];
    cv_g[i]   = (const float*)d_in[base + 1];
    cv_l1W[i] = (const float*)d_in[base + 2];
    cv_l1b[i] = (const float*)d_in[base + 3];
    cv_l2W[i] = (const float*)d_in[base + 4];
    cv_l2b[i] = (const float*)d_in[base + 5];
  }
  const float* gat_W    = (const float*)d_in[33];
  const float* gat_adst = (const float*)d_in[34];
  const float* gat_asrc = (const float*)d_in[35];
  const float* gat_b    = (const float*)d_in[36];
  const float* l1_W = (const float*)d_in[37];
  const float* l1_b = (const float*)d_in[38];
  const float* l2_W = (const float*)d_in[39];
  const float* l2_b = (const float*)d_in[40];
  const int* srcI  = (const int*)d_in[41];
  const int* dstI  = srcI + E;
  const int* batch = (const int*)d_in[42];

  // ---- workspace carve-out ----
  char* wsp = (char*)d_ws;
  size_t off = 0;
  auto alloc = [&](size_t bytes) -> float* {
    float* p = (float*)(wsp + off);
    off = (off + bytes + 255) & ~(size_t)255;
    return p;
  };
  const size_t NF = (size_t)N * FD * sizeof(float);
  float* h   = alloc(NF);
  float* z   = alloc(NF);           // also GAT hh
  float* t0  = alloc(NF);           // also GAT accumulator
  float* t1  = alloc(NF);
  float* mA  = alloc(N * sizeof(float));
  float* mB  = alloc(N * sizeof(float));
  float* z1  = alloc(N * sizeof(float));
  float* a_s = alloc((size_t)N * NHEAD * sizeof(float));
  float* a_d = alloc((size_t)N * NHEAD * sizeof(float));
  float* mx  = alloc((size_t)N * NHEAD * sizeof(float));
  float* den = alloc((size_t)N * NHEAD * sizeof(float));
  float* wexp= alloc((size_t)E * NHEAD * sizeof(float));
  float* deg = alloc(N * sizeof(float));
  float* hs  = alloc(N * sizeof(float));
  float* x2b = alloc(N * sizeof(float));
  float* mean= alloc(FD * sizeof(float));
  float* var = alloc(FD * sizeof(float));
  float* bnsum = alloc(2 * FD * sizeof(float));  // [sums | sqs]
  float* bmax= alloc(G * sizeof(float));
  float* bmin= alloc(G * sizeof(float));
  float* tvol= alloc(G * sizeof(float));
  float* tgt = alloc(G * sizeof(float));
  float* aG  = alloc(G * sizeof(float));
  float* sacc= alloc(2 * G * sizeof(float));   // [sdnk | sxkdk]
  float* racc= alloc(4 * G * sizeof(float));   // [vh | ec1 | tvh | ec2]

  dim3 B(256);
  auto nb = [](long long t) { return dim3((unsigned)((t + 255) / 256)); };
  const dim3 gemmG((unsigned)((N + 127) / 128));
  const long long N16 = (long long)N * 16;   // float4 count over [N,64]
  const long long E16 = (long long)E * 16;

  auto run_bn = [&](float* X, const float* g, const float* b) {
    hipMemsetAsync(bnsum, 0, 2 * FD * sizeof(float), stream);
    bn_partial_kernel<<<64, B, 0, stream>>>((const float4*)X, N16, bnsum, bnsum + FD);
    bn_finalize_kernel<<<1, 64, 0, stream>>>(bnsum, bnsum + FD, N, mean, var);
    bn_apply_kernel<<<nb(N16), B, 0, stream>>>((float4*)X, N16, mean, var, g, b);
  };

  // ---- mask = get_mask(x) ----
  binmask_kernel<<<nb(N), B, 0, stream>>>(x, mA, N);
  maskprop_kernel<<<nb(E), B, 0, stream>>>(x, srcI, dstI, mA, E);

  // ---- conv1 (fin=1) ----
  hipMemcpyAsync(z1, x, N * sizeof(float), hipMemcpyDeviceToDevice, stream);
  scatter1_kernel<<<nb(E), B, 0, stream>>>(x, srcI, dstI, z1, E);
  expand1_kernel<<<nb(N16), B, 0, stream>>>(z1, c1_l1W, c1_l1b, (float4*)t0, N16);
  gemm_n64_wmma_kernel<1, true><<<gemmG, B, 0, stream>>>(t0, c1_l2W, c1_l2b, t1, N);
  run_bn(t1, c1_g, c1_b);
  maskmul_kernel<<<nb(N16), B, 0, stream>>>((const float4*)t1, mA, (float4*)h, N16);

  // ---- 3 residual GIN layers ----
  for (int i = 0; i < 3; ++i) {
    hipMemcpyAsync(z, h, NF, hipMemcpyDeviceToDevice, stream);
    scatter64_kernel<<<nb(E16), B, 0, stream>>>(h, srcI, dstI, z, E);
    gemm_n64_wmma_kernel<1, true><<<gemmG, B, 0, stream>>>(z, cv_l1W[i], cv_l1b[i], t0, N);
    gemm_n64_wmma_kernel<1, true><<<gemmG, B, 0, stream>>>(t0, cv_l2W[i], cv_l2b[i], t1, N);
    run_bn(t1, cv_g[i], cv_b[i]);
    add_kernel<<<nb(N16), B, 0, stream>>>((float4*)h, (const float4*)t1, N16);
    binmask_kernel<<<nb(N), B, 0, stream>>>(mA, mB, N);
    maskprop_kernel<<<nb(E), B, 0, stream>>>(mA, srcI, dstI, mB, E);
    { float* t = mA; mA = mB; mB = t; }
    maskmul_kernel<<<nb(N16), B, 0, stream>>>((const float4*)h, mA, (float4*)h, N16);
    run_bn(h, bns_g[i], bns_b[i]);
  }

  // ---- GAT ----
  gemm_n64_wmma_kernel<0, false><<<gemmG, B, 0, stream>>>(h, gat_W, nullptr, z, N);  // hh
  attn_score_kernel<<<nb(N), B, 0, stream>>>(z, gat_asrc, gat_adst, a_s, a_d, N);
  fill_kernel<<<nb((long long)N * NHEAD), B, 0, stream>>>(mx, -INFINITY, (long long)N * NHEAD);
  hipMemsetAsync(den, 0, (size_t)N * NHEAD * sizeof(float), stream);
  gat_edge1_kernel<<<nb((long long)E * NHEAD), B, 0, stream>>>(a_s, a_d, srcI, dstI, wexp, mx, E);
  gat_edge2_kernel<<<nb((long long)E * NHEAD), B, 0, stream>>>(wexp, mx, dstI, den, E);
  hipMemsetAsync(t0, 0, NF, stream);
  gat_edge3_kernel<<<nb(E16), B, 0, stream>>>(wexp, den, z, srcI, dstI, t0, E);
  binmask_kernel<<<nb(N), B, 0, stream>>>(mA, mB, N);
  maskprop_kernel<<<nb(E), B, 0, stream>>>(mA, srcI, dstI, mB, E);
  { float* t = mA; mA = mB; mB = t; }
  gat_finish_kernel<<<nb(N16), B, 0, stream>>>((const float4*)t0, gat_b, mA, (float4*)h, N16);

  // ---- lin1 / bn2 / lin2 ----
  gemm_n64_wmma_kernel<2, true><<<gemmG, B, 0, stream>>>(h, l1_W, l1_b, t1, N);
  maskmul_kernel<<<nb(N16), B, 0, stream>>>((const float4*)t1, mA, (float4*)t1, N16);
  run_bn(t1, bn2_g, bn2_b);
  lin2_kernel<<<nb(N), B, 0, stream>>>(t1, l2_W, l2_b, mA, hs, N);

  // ---- per-graph min-max normalize, + mask*1e-6 + xinit ----
  fill_kernel<<<nb(G), B, 0, stream>>>(bmax, -INFINITY, G);
  fill_kernel<<<nb(G), B, 0, stream>>>(bmin,  INFINITY, G);
  minmax_kernel<<<nb(N), B, 0, stream>>>(hs, batch, bmax, bmin, N);
  norm_kernel<<<nb(N), B, 0, stream>>>(hs, x, mA, batch, bmax, bmin, N);

  // ---- degrees, volumes, targets ----
  hipMemsetAsync(deg, 0, N * sizeof(float), stream);
  deg_kernel<<<nb(E), B, 0, stream>>>(srcI, deg, E);
  hipMemsetAsync(tvol, 0, G * sizeof(float), stream);
  vol_kernel<<<nb(N), B, 0, stream>>>(deg, batch, tvol, N);
  target_kernel<<<nb(G), B, 0, stream>>>(tvol, tgt, G);

  // ---- 30-iteration cut refinement ----
  fill_kernel<<<nb(G), B, 0, stream>>>(aG, 1.0f, G);
  for (int it = 0; it < 30; ++it) {
    hipMemsetAsync(sacc, 0, 2 * G * sizeof(float), stream);
    refine_node_kernel<<<nb(N), B, 0, stream>>>(hs, deg, mA, batch, aG, sacc, sacc + G, N);
    refine_update_kernel<<<nb(G), B, 0, stream>>>(aG, tgt, sacc, sacc + G, G);
  }

  // ---- outputs: probs | expected_cut | vol_hard | tv_hard ----
  float* probs = (float*)d_out;
  probs_kernel<<<nb(N), B, 0, stream>>>(hs, mA, batch, aG, probs, x2b, N);
  hipMemsetAsync(racc, 0, 4 * G * sizeof(float), stream);
  node_accum_kernel<<<nb(N), B, 0, stream>>>(probs, deg, x2b, batch, racc, racc + G, N);
  edge_accum_kernel<<<nb(E), B, 0, stream>>>(probs, x2b, srcI, dstI, batch, racc + 2*G, racc + 3*G, E);
  finalize_kernel<<<nb(G), B, 0, stream>>>(racc + G, racc + 3*G, racc, racc + 2*G,
                                           probs + N, probs + N + G, probs + N + 2*G, G);
}
